// Qwen2MoeDecoderLayer_283467842488
// MI455X (gfx1250) — matrix operand order repaired and checked by reading, not compiled, not run on hardware
//
#include <hip/hip_runtime.h>
#include <hip/hip_bf16.h>
#include <math.h>

// ---------------- problem constants (match reference) ----------------
#define BDIM 2
#define SDIM 1024
#define HDIM 2048
#define NHQ 16
#define NKVH 4
#define HEADD 128
#define NEXP 16
#define TOPKN 4
#define MIDIM 1408
#define SIDIM 5632
#define MTOK (BDIM * SDIM)               // 2048 tokens
#define QKVN ((NHQ + 2 * NKVH) * HEADD)  // 3072

typedef __bf16 bf16;
typedef __attribute__((ext_vector_type(16))) __bf16 bf16x16;
typedef __attribute__((ext_vector_type(8)))  __bf16 bf16x8;
typedef __attribute__((ext_vector_type(8)))  float  f32x8;

union AFrag { bf16x16 v; bf16x8 h[2]; };

static __device__ __forceinline__ f32x8 wmma_bf16(const AFrag& a, const AFrag& b, f32x8 c) {
  // D = A(16x32 bf16) * B(32x16 bf16) + C(16x16 f32)
  return __builtin_amdgcn_wmma_f32_16x16x32_bf16(false, a.v, false, b.v,
                                                 (short)0, c, false, false);
}

// =====================================================================
// Generic bf16-WMMA GEMM:  out epilogue of  A[M,K]f32 @ W[N,K]f32^T
// MODE 0: out[r,c]  = acc + (bias ? bias[c] : 0)
// MODE 1: out[r,c]  = scale[r*sstride] * acc
// MODE 2: out[r,c] += scale[r*sstride] * acc
// grid = (N/128, M/128), block = 256 (8 waves)
// =====================================================================
#define LDT 40  // LDS row stride in bf16 (32 data + 8 pad) -> 80B rows

template <int MODE>
__global__ __launch_bounds__(256)
void gemm_kernel(const float* __restrict__ A, const float* __restrict__ W,
                 const float* __restrict__ bias,
                 const float* __restrict__ scale, int sstride,
                 float* __restrict__ out, int N, int K) {
  __shared__ bf16 sA[128 * LDT];
  __shared__ bf16 sB[128 * LDT];

  const int tid  = threadIdx.x;
  const int lane = tid & 31;
  const int wid  = tid >> 5;
  const int wr   = wid >> 2;   // 0..1 : wave row (64 rows each)
  const int wc   = wid & 3;    // 0..3 : wave col (32 cols each)
  const int m0   = blockIdx.y * 128;
  const int n0   = blockIdx.x * 128;

  const int lrow = tid >> 1;          // 0..127 tile row loaded by this thread
  const int lcol = (tid & 1) * 16;    // 0 or 16 (16 f32 per thread)

  const int l15 = lane & 15;
  const int c0  = (lane < 16) ? 0 : 8;   // A-frag K sub-chunk base
  const int kb  = (lane < 16) ? 0 : 16;  // unused; B-frag K base is c0 as well

  f32x8 acc[4][2];
#pragma unroll
  for (int mt = 0; mt < 4; ++mt)
#pragma unroll
    for (int nt = 0; nt < 2; ++nt) acc[mt][nt] = (f32x8)0.0f;

  for (int kt = 0; kt < K; kt += 32) {
    {  // stage A tile (f32 -> bf16)
      const float4* s = (const float4*)(A + (size_t)(m0 + lrow) * K + kt + lcol);
      float4 f0 = s[0], f1 = s[1], f2 = s[2], f3 = s[3];
      bf16x8 v0, v1;
      v0[0]=(bf16)f0.x; v0[1]=(bf16)f0.y; v0[2]=(bf16)f0.z; v0[3]=(bf16)f0.w;
      v0[4]=(bf16)f1.x; v0[5]=(bf16)f1.y; v0[6]=(bf16)f1.z; v0[7]=(bf16)f1.w;
      v1[0]=(bf16)f2.x; v1[1]=(bf16)f2.y; v1[2]=(bf16)f2.z; v1[3]=(bf16)f2.w;
      v1[4]=(bf16)f3.x; v1[5]=(bf16)f3.y; v1[6]=(bf16)f3.z; v1[7]=(bf16)f3.w;
      *(bf16x8*)&sA[lrow * LDT + lcol]     = v0;
      *(bf16x8*)&sA[lrow * LDT + lcol + 8] = v1;
    }
    {  // stage W tile (f32 -> bf16)
      const float4* s = (const float4*)(W + (size_t)(n0 + lrow) * K + kt + lcol);
      float4 f0 = s[0], f1 = s[1], f2 = s[2], f3 = s[3];
      bf16x8 v0, v1;
      v0[0]=(bf16)f0.x; v0[1]=(bf16)f0.y; v0[2]=(bf16)f0.z; v0[3]=(bf16)f0.w;
      v0[4]=(bf16)f1.x; v0[5]=(bf16)f1.y; v0[6]=(bf16)f1.z; v0[7]=(bf16)f1.w;
      v1[0]=(bf16)f2.x; v1[1]=(bf16)f2.y; v1[2]=(bf16)f2.z; v1[3]=(bf16)f2.w;
      v1[4]=(bf16)f3.x; v1[5]=(bf16)f3.y; v1[6]=(bf16)f3.z; v1[7]=(bf16)f3.w;
      *(bf16x8*)&sB[lrow * LDT + lcol]     = v0;
      *(bf16x8*)&sB[lrow * LDT + lcol + 8] = v1;
    }
    __syncthreads();

    AFrag af[4], bfm[2];
#pragma unroll
    for (int mt = 0; mt < 4; ++mt) {
      const bf16* p = &sA[(wr * 64 + mt * 16 + l15) * LDT];
      af[mt].h[0] = *(const bf16x8*)(p + c0);
      af[mt].h[1] = *(const bf16x8*)(p + c0 + 16);
    }
#pragma unroll
    for (int nt = 0; nt < 2; ++nt) {
      const bf16* p = &sB[(wc * 32 + nt * 16 + l15) * LDT];
      bfm[nt].h[0] = *(const bf16x8*)(p + c0);
      bfm[nt].h[1] = *(const bf16x8*)(p + c0 + 16);
    }
#pragma unroll
    for (int mt = 0; mt < 4; ++mt)
#pragma unroll
      for (int nt = 0; nt < 2; ++nt)
        acc[mt][nt] = wmma_bf16(af[mt], bfm[nt], acc[mt][nt]);
    __syncthreads();
  }

  // epilogue: C layout -> lane holds col = l15, rows = 8*(lane>>4)+j
#pragma unroll
  for (int mt = 0; mt < 4; ++mt) {
    const int rbase = m0 + wr * 64 + mt * 16 + ((lane >> 4) << 3);
#pragma unroll
    for (int nt = 0; nt < 2; ++nt) {
      const int col = n0 + wc * 32 + nt * 16 + l15;
#pragma unroll
      for (int j = 0; j < 8; ++j) {
        const int row = rbase + j;
        const float v = acc[mt][nt][j];
        const size_t o = (size_t)row * N + col;
        if (MODE == 0) {
          out[o] = v + (bias ? bias[col] : 0.0f);
        } else if (MODE == 1) {
          out[o] = scale[(size_t)row * sstride] * v;
        } else {
          out[o] += scale[(size_t)row * sstride] * v;
        }
      }
    }
  }
}

// =====================================================================
// Flash attention, bf16 WMMA. grid=(S/64, B*NH), block=128 (4 waves).
// Each wave owns 16 query rows; 32-key tiles staged in LDS.
// =====================================================================
__global__ __launch_bounds__(128)
void attn_kernel(const bf16* __restrict__ qh, const bf16* __restrict__ kh,
                 const bf16* __restrict__ vh, float* __restrict__ attn) {
  __shared__ bf16 sK[32 * 136];      // [key][hd], pad 8
  __shared__ bf16 sVt[128 * 40];     // [hd][key], pad 8
  __shared__ bf16 sP[4][16 * 40];    // per-wave P staging [row][key], pad 8

  const int tid  = threadIdx.x;
  const int lane = tid & 31;
  const int wid  = tid >> 5;
  const int l15  = lane & 15;
  const int hi8  = (lane >> 4) << 3;
  const int c0   = (lane < 16) ? 0 : 8;
  const int kb16 = (lane < 16) ? 0 : 16;

  const int q0  = blockIdx.x * 64;
  const int bh  = blockIdx.y;
  const int b   = bh / NHQ;
  const int h   = bh % NHQ;
  const int kvh = h / (NHQ / NKVH);

  const bf16* qbase = qh + ((size_t)(b * NHQ + h) * SDIM + q0 + wid * 16) * HEADD;
  const bf16* kbase = kh + (size_t)(b * NKVH + kvh) * SDIM * HEADD;
  const bf16* vbase = vh + (size_t)(b * NKVH + kvh) * SDIM * HEADD;

  // Q fragments (A-layout) straight from global
  AFrag qf[4];
#pragma unroll
  for (int kc = 0; kc < 4; ++kc) {
    const bf16* p = qbase + (size_t)l15 * HEADD + kc * 32;
    qf[kc].h[0] = *(const bf16x8*)(p + c0);
    qf[kc].h[1] = *(const bf16x8*)(p + c0 + 16);
  }

  float m_i[8], l_i[8];
  f32x8 o[8];
#pragma unroll
  for (int j = 0; j < 8; ++j) { m_i[j] = -1e30f; l_i[j] = 0.0f; }
#pragma unroll
  for (int ct = 0; ct < 8; ++ct) o[ct] = (f32x8)0.0f;

  const int   nkb  = q0 / 32 + 2;                 // key blocks covering q0..q0+63
  const float scl  = 0.088388347648318447f;       // 1/sqrt(128)
  const int   lr   = tid >> 2;                    // 0..31 key row for loads
  const int   lc   = (tid & 3) * 32;              // hd chunk for loads

  for (int kblk = 0; kblk < nkb; ++kblk) {
    __syncthreads();
    {  // K tile -> sK[key][hd]
      const bf16* g = kbase + (size_t)(kblk * 32 + lr) * HEADD + lc;
      bf16* d = &sK[lr * 136 + lc];
      *(bf16x8*)(d)      = *(const bf16x8*)(g);
      *(bf16x8*)(d + 8)  = *(const bf16x8*)(g + 8);
      *(bf16x8*)(d + 16) = *(const bf16x8*)(g + 16);
      *(bf16x8*)(d + 24) = *(const bf16x8*)(g + 24);
    }
    {  // V tile transposed -> sVt[hd][key]
      const bf16* g = vbase + (size_t)(kblk * 32 + lr) * HEADD + lc;
#pragma unroll
      for (int j = 0; j < 32; ++j) sVt[(lc + j) * 40 + lr] = g[j];
    }
    __syncthreads();

    // scores S = Q K^T : two 16x16 tiles
    f32x8 sc[2];
    sc[0] = (f32x8)0.0f; sc[1] = (f32x8)0.0f;
#pragma unroll
    for (int nt = 0; nt < 2; ++nt) {
      const bf16* p = &sK[(nt * 16 + l15) * 136];
#pragma unroll
      for (int kc = 0; kc < 4; ++kc) {
        AFrag bfk;
        bfk.h[0] = *(const bf16x8*)(p + kc * 32 + c0);
        bfk.h[1] = *(const bf16x8*)(p + kc * 32 + c0 + 16);
        sc[nt] = wmma_bf16(qf[kc], bfk, sc[nt]);
      }
    }

    // masked online softmax (per-lane rows hi8+j, reduce across 16-lane half)
    float pv0[8], pv1[8];
#pragma unroll
    for (int j = 0; j < 8; ++j) {
      const int myrow = q0 + wid * 16 + hi8 + j;
      float s0 = sc[0][j] * scl;
      float s1 = sc[1][j] * scl;
      if (kblk * 32 + l15 > myrow)      s0 = -1e30f;
      if (kblk * 32 + 16 + l15 > myrow) s1 = -1e30f;
      float rm = fmaxf(s0, s1);
#pragma unroll
      for (int mk = 8; mk >= 1; mk >>= 1) rm = fmaxf(rm, __shfl_xor(rm, mk, 32));
      const float newm = fmaxf(m_i[j], rm);
      const float corr = __expf(m_i[j] - newm);
      const float p0 = __expf(s0 - newm);
      const float p1 = __expf(s1 - newm);
      float rs = p0 + p1;
#pragma unroll
      for (int mk = 8; mk >= 1; mk >>= 1) rs += __shfl_xor(rs, mk, 32);
      m_i[j] = newm;
      l_i[j] = l_i[j] * corr + rs;
#pragma unroll
      for (int ct = 0; ct < 8; ++ct) o[ct][j] *= corr;
      pv0[j] = p0; pv1[j] = p1;
    }

    // C-layout -> A-frag layout for P via per-wave LDS patch
#pragma unroll
    for (int j = 0; j < 8; ++j) {
      sP[wid][(hi8 + j) * 40 + l15]      = (bf16)pv0[j];
      sP[wid][(hi8 + j) * 40 + 16 + l15] = (bf16)pv1[j];
    }
    AFrag pf;
    {
      const bf16* p = &sP[wid][l15 * 40];
      pf.h[0] = *(const bf16x8*)(p + c0);
      pf.h[1] = *(const bf16x8*)(p + c0 + 16);
    }

    // O += P V  (8 col tiles of 16)
#pragma unroll
    for (int ct = 0; ct < 8; ++ct) {
      AFrag vf;
      const bf16* p = &sVt[(ct * 16 + l15) * 40 + c0];
      vf.h[0] = *(const bf16x8*)(p);
      vf.h[1] = *(const bf16x8*)(p + 16);
      o[ct] = wmma_bf16(pf, vf, o[ct]);
    }
  }

  // store [B,S,NH*HD] as f32
  const int trow = q0 + wid * 16 + hi8;
#pragma unroll
  for (int j = 0; j < 8; ++j) {
    const float inv_l = 1.0f / l_i[j];
    const size_t ro = ((size_t)b * SDIM + trow + j) * (NHQ * HEADD) + h * HEADD;
#pragma unroll
    for (int ct = 0; ct < 8; ++ct)
      attn[ro + ct * 16 + l15] = o[ct][j] * inv_l;
  }
}

// =====================================================================
// Elementwise / reduction helpers
// =====================================================================
__global__ __launch_bounds__(256)
void rmsnorm_kernel(const float* __restrict__ x, const float* __restrict__ w,
                    float* __restrict__ out) {
  __shared__ float red[256];
  const int row = blockIdx.x, tid = threadIdx.x;
  float v[8], s = 0.0f;
#pragma unroll
  for (int k = 0; k < 8; ++k) {
    v[k] = x[(size_t)row * HDIM + tid + k * 256];
    s += v[k] * v[k];
  }
  red[tid] = s; __syncthreads();
  for (int off = 128; off > 0; off >>= 1) {
    if (tid < off) red[tid] += red[tid + off];
    __syncthreads();
  }
  const float rstd = rsqrtf(red[0] / (float)HDIM + 1e-6f);
#pragma unroll
  for (int k = 0; k < 8; ++k) {
    const int i = tid + k * 256;
    out[(size_t)row * HDIM + i] = v[k] * rstd * w[i];
  }
}

__global__ __launch_bounds__(256)
void add_rmsnorm_kernel(const float* __restrict__ a, const float* __restrict__ b,
                        const float* __restrict__ w, float* __restrict__ resid,
                        float* __restrict__ out) {
  __shared__ float red[256];
  const int row = blockIdx.x, tid = threadIdx.x;
  float v[8], s = 0.0f;
#pragma unroll
  for (int k = 0; k < 8; ++k) {
    const int i = tid + k * 256;
    const float r = a[(size_t)row * HDIM + i] + b[(size_t)row * HDIM + i];
    v[k] = r;
    resid[(size_t)row * HDIM + i] = r;
    s += r * r;
  }
  red[tid] = s; __syncthreads();
  for (int off = 128; off > 0; off >>= 1) {
    if (tid < off) red[tid] += red[tid + off];
    __syncthreads();
  }
  const float rstd = rsqrtf(red[0] / (float)HDIM + 1e-6f);
#pragma unroll
  for (int k = 0; k < 8; ++k) {
    const int i = tid + k * 256;
    out[(size_t)row * HDIM + i] = v[k] * rstd * w[i];
  }
}

__global__ __launch_bounds__(256)
void rope_kernel(const float* __restrict__ qkv, const int* __restrict__ pos,
                 bf16* __restrict__ qh, bf16* __restrict__ kh) {
  const int idx = blockIdx.x * 256 + threadIdx.x;       // over MTOK*(NHQ+NKVH)*64
  const int i    = idx & 63;
  const int p    = idx >> 6;
  const int hidx = p % (NHQ + NKVH);
  const int tok  = p / (NHQ + NKVH);
  const int b = tok / SDIM, s = tok % SDIM;
  const float inv = __powf(1000000.0f, -(float)i * (1.0f / 64.0f));
  const float f = (float)pos[tok] * inv;
  float sn, cs;
  __sincosf(f, &sn, &cs);
  const float* src; bf16* dst;
  if (hidx < NHQ) {
    src = qkv + (size_t)tok * QKVN + hidx * HEADD;
    dst = qh + ((size_t)(b * NHQ + hidx) * SDIM + s) * HEADD;
  } else {
    const int h2 = hidx - NHQ;
    src = qkv + (size_t)tok * QKVN + NHQ * HEADD + h2 * HEADD;
    dst = kh + ((size_t)(b * NKVH + h2) * SDIM + s) * HEADD;
  }
  const float x1 = src[i], x2 = src[i + 64];
  dst[i]      = (bf16)(x1 * cs - x2 * sn);
  dst[i + 64] = (bf16)(x2 * cs + x1 * sn);
}

__global__ __launch_bounds__(256)
void vcopy_kernel(const float* __restrict__ qkv, bf16* __restrict__ vh) {
  const int idx = blockIdx.x * 256 + threadIdx.x;  // over MTOK*NKVH*HEADD
  const int d  = idx & 127;
  const int p  = idx >> 7;
  const int hv = p % NKVH;
  const int tok = p / NKVH;
  const int b = tok / SDIM, s = tok % SDIM;
  vh[((size_t)(b * NKVH + hv) * SDIM + s) * HEADD + d] =
      (bf16)qkv[(size_t)tok * QKVN + (NHQ + NKVH) * HEADD + hv * HEADD + d];
}

__global__ __launch_bounds__(256)
void silu_mul_kernel(const float* __restrict__ g, float* __restrict__ out, int Ninner) {
  const size_t idx = (size_t)blockIdx.x * 256 + threadIdx.x;
  const size_t row = idx / Ninner;
  const int    i   = (int)(idx % Ninner);
  const float a = g[row * (size_t)(2 * Ninner) + i];
  const float b = g[row * (size_t)(2 * Ninner) + Ninner + i];
  out[idx] = (a / (1.0f + __expf(-a))) * b;
}

__global__ __launch_bounds__(256)
void router_kernel(const float* __restrict__ x, const float* __restrict__ gw,
                   const float* __restrict__ sgw, float* __restrict__ combine,
                   float* __restrict__ sgate) {
  __shared__ float red[256];
  __shared__ float logit[17];
  const int row = blockIdx.x, tid = threadIdx.x;
  float acc[17];
#pragma unroll
  for (int e = 0; e < 17; ++e) acc[e] = 0.0f;
  for (int i = tid; i < HDIM; i += 256) {
    const float xv = x[(size_t)row * HDIM + i];
#pragma unroll
    for (int e = 0; e < 16; ++e) acc[e] += xv * gw[e * HDIM + i];
    acc[16] += xv * sgw[i];
  }
  for (int e = 0; e < 17; ++e) {
    red[tid] = acc[e]; __syncthreads();
    for (int off = 128; off > 0; off >>= 1) {
      if (tid < off) red[tid] += red[tid + off];
      __syncthreads();
    }
    if (tid == 0) logit[e] = red[0];
    __syncthreads();
  }
  if (tid == 0) {
    float mx = logit[0];
    for (int e = 1; e < 16; ++e) mx = fmaxf(mx, logit[e]);
    float pr[16], sum = 0.0f;
    for (int e = 0; e < 16; ++e) { pr[e] = __expf(logit[e] - mx); sum += pr[e]; }
    const float inv = 1.0f / sum;
    for (int e = 0; e < 16; ++e) pr[e] *= inv;
    float outw[16];
    for (int e = 0; e < 16; ++e) outw[e] = 0.0f;
    for (int t = 0; t < TOPKN; ++t) {
      int best = 0; float bv = -1.0f;
      for (int e = 0; e < 16; ++e)
        if (pr[e] > bv) { bv = pr[e]; best = e; }
      outw[best] = bv; pr[best] = -2.0f;
    }
    for (int e = 0; e < 16; ++e) combine[(size_t)row * 16 + e] = outw[e];
    sgate[row] = 1.0f / (1.0f + __expf(-logit[16]));
  }
}

// =====================================================================
// Orchestration
// =====================================================================
extern "C" void kernel_launch(void* const* d_in, const int* in_sizes, int n_in,
                              void* d_out, int out_size, void* d_ws, size_t ws_size,
                              hipStream_t stream) {
  const int*   positions = (const int*)  d_in[0];
  const float* hidden    = (const float*)d_in[1];
  const float* in_ln_w   = (const float*)d_in[2];
  const float* post_ln_w = (const float*)d_in[3];
  const float* qkv_w     = (const float*)d_in[4];
  const float* qkv_b     = (const float*)d_in[5];
  const float* o_w       = (const float*)d_in[6];
  const float* gate_w    = (const float*)d_in[7];
  const float* sgw       = (const float*)d_in[8];
  const float* segu_w    = (const float*)d_in[9];
  const float* sedown_w  = (const float*)d_in[10];
  const float* w13       = (const float*)d_in[11];
  const float* w2        = (const float*)d_in[12];

  char* ws = (char*)d_ws;
  size_t off = 0;
  auto alloc = [&](size_t bytes) -> char* {
    char* p = ws + off;
    off += (bytes + 255) & ~(size_t)255;
    return p;
  };
  float* hn      = (float*)alloc((size_t)MTOK * HDIM * 4);
  float* qkv     = (float*)alloc((size_t)MTOK * QKVN * 4);
  bf16*  qh      = (bf16*) alloc((size_t)MTOK * NHQ * HEADD * 2);
  bf16*  kh      = (bf16*) alloc((size_t)MTOK * NKVH * HEADD * 2);
  bf16*  vh      = (bf16*) alloc((size_t)MTOK * NKVH * HEADD * 2);
  float* attn    = (float*)alloc((size_t)MTOK * HDIM * 4);
  float* oout    = (float*)alloc((size_t)MTOK * HDIM * 4);
  float* xbuf    = (float*)alloc((size_t)MTOK * HDIM * 4);
  float* gbuf    = (float*)alloc((size_t)MTOK * 2 * SIDIM * 4);  // reused by experts
  float* abuf    = (float*)alloc((size_t)MTOK * SIDIM * 4);      // reused by experts
  float* combine = (float*)alloc((size_t)MTOK * NEXP * 4);
  float* sgate   = (float*)alloc((size_t)MTOK * 4);

  float* moe_out = (float*)d_out;
  float* resid   = moe_out + (size_t)MTOK * HDIM;

  // 1) input RMSNorm
  rmsnorm_kernel<<<MTOK, 256, 0, stream>>>(hidden, in_ln_w, hn);
  // 2) QKV projection (+bias)
  gemm_kernel<0><<<dim3(QKVN / 128, MTOK / 128), 256, 0, stream>>>(
      hn, qkv_w, qkv_b, nullptr, 0, qkv, QKVN, HDIM);
  // 3) RoPE + head-major bf16 repack
  rope_kernel<<<(MTOK * (NHQ + NKVH) * 64) / 256, 256, 0, stream>>>(qkv, positions, qh, kh);
  vcopy_kernel<<<(MTOK * NKVH * HEADD) / 256, 256, 0, stream>>>(qkv, vh);
  // 4) causal flash attention
  attn_kernel<<<dim3(SDIM / 64, BDIM * NHQ), 128, 0, stream>>>(qh, kh, vh, attn);
  // 5) output projection
  gemm_kernel<0><<<dim3(HDIM / 128, MTOK / 128), 256, 0, stream>>>(
      attn, o_w, nullptr, nullptr, 0, oout, HDIM, HDIM);
  // 6) residual add + post RMSNorm
  add_rmsnorm_kernel<<<MTOK, 256, 0, stream>>>(oout, hidden, post_ln_w, resid, xbuf);
  // 7) router (softmax + top-4) and shared-expert sigmoid gate
  router_kernel<<<MTOK, 256, 0, stream>>>(xbuf, gate_w, sgw, combine, sgate);
  // 8) shared expert: gate_up -> silu*mul -> down (assign sgate-scaled into moe_out)
  gemm_kernel<0><<<dim3((2 * SIDIM) / 128, MTOK / 128), 256, 0, stream>>>(
      xbuf, segu_w, nullptr, nullptr, 0, gbuf, 2 * SIDIM, HDIM);
  silu_mul_kernel<<<(unsigned)(((size_t)MTOK * SIDIM) / 256), 256, 0, stream>>>(
      gbuf, abuf, SIDIM);
  gemm_kernel<1><<<dim3(HDIM / 128, MTOK / 128), 256, 0, stream>>>(
      abuf, sedown_w, nullptr, sgate, 1, moe_out, HDIM, SIDIM);
  // 9) dense expert loop (matches reference semantics; combine==0 for unselected)
  for (int e = 0; e < NEXP; ++e) {
    const float* w13e = w13 + (size_t)e * 2 * MIDIM * HDIM;
    const float* w2e  = w2  + (size_t)e * HDIM * MIDIM;
    gemm_kernel<0><<<dim3((2 * MIDIM) / 128, MTOK / 128), 256, 0, stream>>>(
        xbuf, w13e, nullptr, nullptr, 0, gbuf, 2 * MIDIM, HDIM);
    silu_mul_kernel<<<(unsigned)(((size_t)MTOK * MIDIM) / 256), 256, 0, stream>>>(
        gbuf, abuf, MIDIM);
    gemm_kernel<2><<<dim3(HDIM / 128, MTOK / 128), 256, 0, stream>>>(
        abuf, w2e, nullptr, combine + e, NEXP, moe_out, HDIM, MIDIM);
  }
  (void)in_sizes; (void)n_in; (void)out_size; (void)ws_size;
}
